// MultiHeadAttentionBlock_52192442581113
// MI455X (gfx1250) — matrix operand order repaired and checked by reading, not compiled, not run on hardware
//
#include <hip/hip_runtime.h>
#include <hip/hip_bf16.h>

// ---------------------------------------------------------------------------
// MHA block for MI455X (gfx1250, wave32, WMMA, async-to-LDS).
//   D=2048, H=16, DK=128, B=2, S=2048
// Pipeline:
//   1) f32 -> bf16 convert of q,k,v and Wq,Wk,Wv,Wo
//   2) WMMA GEMM (async-LDS weight staging):
//        Q = q@Wq^T+bq, K = k@Wk^T+bk (bf16 out, natural layout)
//        V = v@Wv^T+bv (bf16 out, per-head transposed [b][h][dk][s])
//   3) flash-attention (online softmax), bf16 out natural layout
//   4) WMMA GEMM: out = attn@Wo^T+bo (f32 out)
// ---------------------------------------------------------------------------

typedef __attribute__((ext_vector_type(16))) __bf16 v16bf;
typedef __attribute__((ext_vector_type(8)))  __bf16 v8bf;
typedef __attribute__((ext_vector_type(8)))  float  v8f;
typedef __attribute__((ext_vector_type(4)))  int    v4i;

static constexpr int kB  = 2;
static constexpr int kS  = 2048;
static constexpr int kD  = 2048;
static constexpr int kH  = 16;
static constexpr int kDK = 128;

union V16U { v16bf v; v8bf h[2]; };

#if __has_builtin(__builtin_amdgcn_global_load_async_to_lds_b128)
#define HAVE_ASYNC_LDS 1
#else
#define HAVE_ASYNC_LDS 0
#endif

__device__ __forceinline__ unsigned short f2bf(float x) {
  unsigned int u = __float_as_uint(x);
  u += 0x7FFFu + ((u >> 16) & 1u);   // round-to-nearest-even
  return (unsigned short)(u >> 16);
}

__device__ __forceinline__ v8f v8f_zero() {
  v8f v;
  #pragma unroll
  for (int i = 0; i < 8; ++i) v[i] = 0.0f;
  return v;
}

// Async (or fallback sync) 16-byte global -> LDS copy.
__device__ __forceinline__ void copy16_g2l(const unsigned short* g,
                                           unsigned short* l) {
#if HAVE_ASYNC_LDS
  typedef __attribute__((address_space(1))) v4i gv4i;
  typedef __attribute__((address_space(3))) v4i lv4i;
  __builtin_amdgcn_global_load_async_to_lds_b128((gv4i*)g, (lv4i*)l, 0, 0);
#else
  *(v8bf*)l = *(const v8bf*)g;
#endif
}

__device__ __forceinline__ void wait_async_le2() {
#if HAVE_ASYNC_LDS
#if __has_builtin(__builtin_amdgcn_s_wait_asynccnt)
  __builtin_amdgcn_s_wait_asynccnt(2);
#else
  asm volatile("s_wait_asynccnt 0x2" ::: "memory");
#endif
#endif
}

__device__ __forceinline__ void wait_async_le0() {
#if HAVE_ASYNC_LDS
#if __has_builtin(__builtin_amdgcn_s_wait_asynccnt)
  __builtin_amdgcn_s_wait_asynccnt(0);
#else
  asm volatile("s_wait_asynccnt 0x0" ::: "memory");
#endif
#endif
}

// Load a 16x32 bf16 A-fragment chunk: ptr already includes (lane>>4)*16 offset;
// elements at +0..15 (16 contiguous k-values per lane, matching B).
__device__ __forceinline__ v16bf load_afrag(const unsigned short* p) {
  V16U u;
  u.h[0] = *(const v8bf*)(p);
  u.h[1] = *(const v8bf*)(p + 8);
  return u.v;
}

// Load a 32x16 bf16 B-fragment (16 contiguous k-values per lane).
__device__ __forceinline__ v16bf load_bfrag(const unsigned short* p) {
  V16U u;
  u.h[0] = *(const v8bf*)(p);
  u.h[1] = *(const v8bf*)(p + 8);
  return u.v;
}

__device__ __forceinline__ v8f wmma_bf16(v16bf a, v16bf b, v8f c) {
  return __builtin_amdgcn_wmma_f32_16x16x32_bf16(false, a, false, b,
                                                 (short)0, c, false, false);
}

// ---------------------------------------------------------------------------
// f32 -> bf16 conversion
// ---------------------------------------------------------------------------
__global__ void cvt_f32_bf16_kernel(const float* __restrict__ in,
                                    unsigned short* __restrict__ out,
                                    long long n) {
  long long i = (long long)blockIdx.x * blockDim.x + threadIdx.x;
  long long stride = (long long)gridDim.x * blockDim.x;
  for (; i < n; i += stride) out[i] = f2bf(in[i]);
}

// ---------------------------------------------------------------------------
// GEMM: C[M,N] = A[M,K] @ B[N,K]^T + bias[N]
// A,B bf16 row-major. MODE 0: bf16 natural, 1: bf16 per-head-transposed (V),
// 2: f32 natural (final output).
// Block = 256 threads (8 waves, 4x2), block tile 128x128, wave tile 32x64.
// Weight tile (128x32 bf16) is double-buffered in LDS via async-to-LDS copies
// tracked by ASYNCcnt; A fragments stream direct from global (L2-resident).
// ---------------------------------------------------------------------------
static constexpr int kBRow = 40;  // padded LDS row stride (elems) for 64 banks

template <int MODE>
__global__ __launch_bounds__(256) void gemm_wmma_kernel(
    const unsigned short* __restrict__ A, const unsigned short* __restrict__ Bw,
    const float* __restrict__ bias, void* __restrict__ Out,
    int M, int N, int K) {
  __shared__ __align__(16) unsigned short ldsB[2][128 * kBRow];  // 20 KB

  const int tid  = threadIdx.x;
  const int lane = tid & 31;
  const int wave = tid >> 5;
  const int wm = wave >> 1;  // 0..3
  const int wn = wave & 1;   // 0..1
  const int nblk = blockIdx.x * 128;
  const int m0 = blockIdx.y * 128 + wm * 32;
  const int n0 = nblk + wn * 64;
  const int lr = lane & 15;
  const int hi = lane >> 4;

  const unsigned short* arow0 = A + (size_t)(m0 + lr) * K + hi * 16;
  const unsigned short* arow1 = arow0 + (size_t)16 * K;

  // Stage issue: copy 128 rows x 32 k (8 KB) of weights into ldsB[buf].
  // 512 chunks of 16B; each of 256 threads issues 2 async copies.
  auto issue_stage = [&](int s, int buf) {
    #pragma unroll
    for (int u = 0; u < 2; ++u) {
      const int c  = tid + u * 256;
      const int r  = c >> 2;
      const int cc = c & 3;
      const unsigned short* gp =
          Bw + (size_t)(nblk + r) * K + s * 32 + cc * 8;
      unsigned short* lp = &ldsB[buf][r * kBRow + cc * 8];
      copy16_g2l(gp, lp);
    }
  };

  v8f acc[2][4];
  #pragma unroll
  for (int i = 0; i < 2; ++i)
    #pragma unroll
    for (int j = 0; j < 4; ++j) acc[i][j] = v8f_zero();

  const int NS = K / 32;
  issue_stage(0, 0);
  issue_stage(1, 1);

  for (int s = 0; s < NS; ++s) {
    if (s + 1 < NS) wait_async_le2(); else wait_async_le0();
    __syncthreads();

    const int kt = s * 32;
    v16bf a0 = load_afrag(arow0 + kt);
    v16bf a1 = load_afrag(arow1 + kt);

    const unsigned short* lb = ldsB[s & 1];
    v16bf bf[4];
    #pragma unroll
    for (int j = 0; j < 4; ++j)
      bf[j] = load_bfrag(lb + (wn * 64 + j * 16 + lr) * kBRow + hi * 16);

    #pragma unroll
    for (int j = 0; j < 4; ++j) {
      acc[0][j] = wmma_bf16(a0, bf[j], acc[0][j]);
      acc[1][j] = wmma_bf16(a1, bf[j], acc[1][j]);
    }

    __syncthreads();
    if (s + 2 < NS) issue_stage(s + 2, s & 1);
  }

  // Epilogue. C layout: lanes 0-15 hold rows e(0..7), lanes 16-31 rows e+8.
  #pragma unroll
  for (int j = 0; j < 4; ++j) {
    const int col = n0 + j * 16 + lr;
    const float bv = bias[col];
    #pragma unroll
    for (int i = 0; i < 2; ++i) {
      const int rbase = m0 + i * 16 + hi * 8;
      #pragma unroll
      for (int e = 0; e < 8; ++e) {
        const int row = rbase + e;
        const float v = acc[i][j][e] + bv;
        if (MODE == 0) {
          ((unsigned short*)Out)[(size_t)row * N + col] = f2bf(v);
        } else if (MODE == 1) {
          // V output: [b][h][dk][s], row = b*S+s, col = h*DK+dk
          const int bb = row / kS, ss = row % kS;
          const int hh = col / kDK, dk = col % kDK;
          ((unsigned short*)Out)[(((size_t)bb * kH + hh) * kDK + dk) * kS + ss] =
              f2bf(v);
        } else {
          ((float*)Out)[(size_t)row * N + col] = v;
        }
      }
    }
  }
}

// ---------------------------------------------------------------------------
// Flash attention: one wave handles 16 query rows for one (b,h).
// Loops over 32-key tiles: 8 WMMA (Q.K^T) + 8 WMMA (P.V) per tile.
// ---------------------------------------------------------------------------
__global__ __launch_bounds__(128) void attn_wmma_kernel(
    const unsigned short* __restrict__ Q, const unsigned short* __restrict__ Kp,
    const unsigned short* __restrict__ Vt, unsigned short* __restrict__ Oout) {
  __shared__ __align__(16) unsigned short ldsP[4][16 * 32];

  const int lane = threadIdx.x & 31;
  const int wv = threadIdx.x >> 5;
  const int gw = blockIdx.x * 4 + wv;
  const int qt = gw & (kS / 16 - 1);      // 128 q-tiles
  const int tmp = gw >> 7;
  const int h = tmp & (kH - 1);
  const int b = tmp >> 4;
  const int q0 = qt * 16;
  const int lr = lane & 15;
  const int hi = lane >> 4;

  // Preload Q A-fragments (16 rows x 128 features = 4 K-steps of 32).
  const unsigned short* qrow =
      Q + ((size_t)(b * kS + q0 + lr) * kD + h * kDK) + hi * 16;
  v16bf qa[4];
  #pragma unroll
  for (int st = 0; st < 4; ++st) qa[st] = load_afrag(qrow + st * 32);

  const unsigned short* kbase = Kp + (size_t)(b * kS) * kD + h * kDK;
  const unsigned short* vbase = Vt + ((size_t)(b * kH + h)) * kDK * kS;

  const float scale = 0.08838834764831845f;  // 1/sqrt(128)

  float mrow[8], lrow[8];
  v8f o[8];
  #pragma unroll
  for (int e = 0; e < 8; ++e) { mrow[e] = -1e30f; lrow[e] = 0.0f; }
  #pragma unroll
  for (int ft = 0; ft < 8; ++ft) o[ft] = v8f_zero();

  unsigned short* pl = ldsP[wv];

  for (int k0 = 0; k0 < kS; k0 += 32) {
    // --- scores: S[16 x 32] = Q(16x128) . K(32x128)^T ---
    v8f s0 = v8f_zero(), s1 = v8f_zero();
    const unsigned short* kr0 = kbase + (size_t)(k0 + lr) * kD + hi * 16;
    const unsigned short* kr1 = kr0 + (size_t)16 * kD;
    #pragma unroll
    for (int st = 0; st < 4; ++st) {
      v16bf b0 = *(const v16bf*)(kr0 + st * 32);
      v16bf b1 = *(const v16bf*)(kr1 + st * 32);
      s0 = wmma_bf16(qa[st], b0, s0);
      s1 = wmma_bf16(qa[st], b1, s1);
    }

    // --- online softmax update (rows live in 16-lane halves) ---
    float corr[8];
    #pragma unroll
    for (int e = 0; e < 8; ++e) {
      float x0 = s0[e] * scale;
      float x1 = s1[e] * scale;
      float t = fmaxf(x0, x1);
      t = fmaxf(t, __shfl_xor(t, 1, 32));
      t = fmaxf(t, __shfl_xor(t, 2, 32));
      t = fmaxf(t, __shfl_xor(t, 4, 32));
      t = fmaxf(t, __shfl_xor(t, 8, 32));
      const float nm = fmaxf(mrow[e], t);
      const float p0 = __expf(x0 - nm);
      const float p1 = __expf(x1 - nm);
      float rs = p0 + p1;
      rs += __shfl_xor(rs, 1, 32);
      rs += __shfl_xor(rs, 2, 32);
      rs += __shfl_xor(rs, 4, 32);
      rs += __shfl_xor(rs, 8, 32);
      corr[e] = __expf(mrow[e] - nm);
      lrow[e] = lrow[e] * corr[e] + rs;
      mrow[e] = nm;
      const int row = e + hi * 8;
      pl[row * 32 + lr] = f2bf(p0);
      pl[row * 32 + 16 + lr] = f2bf(p1);
    }

    // rescale output accumulator by per-row correction
    #pragma unroll
    for (int ft = 0; ft < 8; ++ft)
      #pragma unroll
      for (int e = 0; e < 8; ++e) o[ft][e] *= corr[e];

    // reload P as A-fragment (LDS ops are in-order within the wave)
    v16bf pa = load_afrag(pl + lr * 32 + hi * 16);

    // --- O += P(16x32) . V(32x128) ---
    #pragma unroll
    for (int ft = 0; ft < 8; ++ft) {
      const unsigned short* vr =
          vbase + (size_t)(ft * 16 + lr) * kS + k0 + hi * 16;
      v16bf bvf = *(const v16bf*)(vr);
      o[ft] = wmma_bf16(pa, bvf, o[ft]);
    }
  }

  // normalize and store bf16 natural layout [B,S,D]
  float inv[8];
  #pragma unroll
  for (int e = 0; e < 8; ++e) inv[e] = 1.0f / lrow[e];
  #pragma unroll
  for (int ft = 0; ft < 8; ++ft) {
    const int col = h * kDK + ft * 16 + lr;
    #pragma unroll
    for (int e = 0; e < 8; ++e) {
      const int row = q0 + e + hi * 8;
      Oout[(size_t)(b * kS + row) * kD + col] = f2bf(o[ft][e] * inv[e]);
    }
  }
}

// ---------------------------------------------------------------------------
// Host-side launcher
// ---------------------------------------------------------------------------
extern "C" void kernel_launch(void* const* d_in, const int* in_sizes, int n_in,
                              void* d_out, int out_size, void* d_ws,
                              size_t ws_size, hipStream_t stream) {
  (void)in_sizes; (void)n_in; (void)out_size; (void)ws_size;

  const float* q  = (const float*)d_in[0];
  const float* k  = (const float*)d_in[1];
  const float* v  = (const float*)d_in[2];
  // d_in[3] = mask (python scalar 0) -> masking disabled in reference path
  const float* Wq = (const float*)d_in[4];
  const float* bq = (const float*)d_in[5];
  const float* Wk = (const float*)d_in[6];
  const float* bk = (const float*)d_in[7];
  const float* Wv = (const float*)d_in[8];
  const float* bv = (const float*)d_in[9];
  const float* Wo = (const float*)d_in[10];
  const float* bo = (const float*)d_in[11];
  float* out = (float*)d_out;

  const size_t nBSD = (size_t)kB * kS * kD;  // 8,388,608
  const size_t nDD  = (size_t)kD * kD;       // 4,194,304

  unsigned short* ws = (unsigned short*)d_ws;
  unsigned short* qb   = ws;            // bf16 inputs
  unsigned short* kb   = qb + nBSD;
  unsigned short* vb   = kb + nBSD;
  unsigned short* wqb  = vb + nBSD;     // bf16 weights
  unsigned short* wkb  = wqb + nDD;
  unsigned short* wvb  = wkb + nDD;
  unsigned short* wob  = wvb + nDD;
  unsigned short* Qp   = wob + nDD;     // projected Q (bf16, [B,S,D])
  unsigned short* Kpj  = Qp + nBSD;     // projected K (bf16, [B,S,D])
  unsigned short* VtB  = Kpj + nBSD;    // projected V (bf16, [b][h][dk][s])
  unsigned short* attn = VtB + nBSD;    // attention output (bf16, [B,S,D])

  const int cvtBlk = 256;
  const int cvtGrid = 2048;
  cvt_f32_bf16_kernel<<<cvtGrid, cvtBlk, 0, stream>>>(q, qb, (long long)nBSD);
  cvt_f32_bf16_kernel<<<cvtGrid, cvtBlk, 0, stream>>>(k, kb, (long long)nBSD);
  cvt_f32_bf16_kernel<<<cvtGrid, cvtBlk, 0, stream>>>(v, vb, (long long)nBSD);
  cvt_f32_bf16_kernel<<<cvtGrid, cvtBlk, 0, stream>>>(Wq, wqb, (long long)nDD);
  cvt_f32_bf16_kernel<<<cvtGrid, cvtBlk, 0, stream>>>(Wk, wkb, (long long)nDD);
  cvt_f32_bf16_kernel<<<cvtGrid, cvtBlk, 0, stream>>>(Wv, wvb, (long long)nDD);
  cvt_f32_bf16_kernel<<<cvtGrid, cvtBlk, 0, stream>>>(Wo, wob, (long long)nDD);

  const int M = kB * kS;  // 4096
  const int N = kD;       // 2048
  const int K = kD;       // 2048
  dim3 gblk(256);
  dim3 ggrid(N / 128, M / 128);  // (16, 32)

  gemm_wmma_kernel<0><<<ggrid, gblk, 0, stream>>>(qb, wqb, bq, Qp, M, N, K);
  gemm_wmma_kernel<0><<<ggrid, gblk, 0, stream>>>(kb, wkb, bk, Kpj, M, N, K);
  gemm_wmma_kernel<1><<<ggrid, gblk, 0, stream>>>(vb, wvb, bv, VtB, M, N, K);

  // B*H*(S/16) = 4096 wave-tasks, 4 waves per 128-thread block -> 1024 blocks
  attn_wmma_kernel<<<1024, 128, 0, stream>>>(Qp, Kpj, VtB, attn);

  gemm_wmma_kernel<2><<<ggrid, gblk, 0, stream>>>(attn, wob, bo, out, M, N, K);
}
